// CausalSelfAttention_41154376631050
// MI455X (gfx1250) — matrix operand order repaired and checked by reading, not compiled, not run on hardware
//
#include <hip/hip_runtime.h>
#include <hip/hip_bf16.h>
#include <stdint.h>

// ---------------------------------------------------------------------------
// CDNA5 (gfx1250) wave32 WMMA bf16 path.
// ---------------------------------------------------------------------------
typedef __attribute__((ext_vector_type(16))) __bf16 v16bf;
typedef __attribute__((ext_vector_type(8)))  float  v8f;

union FragB16 { v16bf v; uint32_t u[8]; };
union FragF32 { v8f v; float f[8]; };

__device__ __forceinline__ uint16_t f2bf(float f) {
  uint32_t u = __builtin_bit_cast(uint32_t, f);
  u += 0x7FFFu + ((u >> 16) & 1u);   // round-to-nearest-even
  return (uint16_t)(u >> 16);
}
__device__ __forceinline__ uint32_t pack2bf(float lo, float hi) {
  return (uint32_t)f2bf(lo) | ((uint32_t)f2bf(hi) << 16);
}

__device__ __forceinline__ v8f wmma_bf16(const FragB16& a, const FragB16& b, v8f c) {
  // D = A(16x32 bf16) * B(32x16 bf16) + C(16x16 f32)
  return __builtin_amdgcn_wmma_f32_16x16x32_bf16(
      /*neg_a=*/false, a.v, /*neg_b=*/false, b.v,
      /*c_mod=*/(short)0, c, /*reuse_a=*/false, /*reuse_b=*/false);
}

// Problem constants
#define BB 4
#define TT 2048
#define CC 1024
#define HH 16
#define DD 64
#define MROWS (BB * TT)          // 8192
#define N_QKV (3 * CC)           // 3072

// ---------------------------------------------------------------------------
// Prep kernel A: flat f32 -> bf16 conversion (x). One float4 per thread.
// ---------------------------------------------------------------------------
__global__ __launch_bounds__(256) void convert_bf16_kernel(
    const float* __restrict__ src, uint16_t* __restrict__ dst, int n4)
{
  int idx = blockIdx.x * 256 + threadIdx.x;
  if (idx < n4) {
    const float4 f = ((const float4*)src)[idx];
    uint2 o;
    o.x = pack2bf(f.x, f.y);
    o.y = pack2bf(f.z, f.w);
    ((uint2*)dst)[idx] = o;
  }
}

// ---------------------------------------------------------------------------
// Prep kernel B: W[K,N] f32 -> Wt[N,K] bf16 (transpose + convert), 32x32 tiles.
// Makes GEMM B-staging a contiguous bf16 copy identical to A-staging.
// ---------------------------------------------------------------------------
__global__ __launch_bounds__(256) void transpose_convert_kernel(
    const float* __restrict__ W, uint16_t* __restrict__ Wt, int K, int N)
{
  __shared__ uint16_t tile[32][33];
  const int tid = threadIdx.x;
  const int k0 = blockIdx.y * 32;
  const int n0 = blockIdx.x * 32;
#pragma unroll
  for (int i = 0; i < 4; i++) {
    int lin = tid + i * 256;
    int kl = lin >> 5, nl = lin & 31;
    tile[nl][kl] = f2bf(W[(size_t)(k0 + kl) * N + n0 + nl]);
  }
  __syncthreads();
#pragma unroll
  for (int i = 0; i < 2; i++) {
    int lin = tid + i * 256;              // 512 u32 writes
    int nl = lin >> 4, k2 = (lin & 15) * 2;
    uint32_t v = (uint32_t)tile[nl][k2] | ((uint32_t)tile[nl][k2 + 1] << 16);
    *(uint32_t*)&Wt[(size_t)(n0 + nl) * K + k0 + k2] = v;
  }
}

// ---------------------------------------------------------------------------
// Kernel 1: QKV GEMM.  y[8192,3072] = Xbf[8192,1024] @ Wt[3072,1024]^T
// 256 threads = 8 waves; block tile 128x128, K-step 32.
// Wave (wr,wc) computes 64x32 (4x2 WMMA tiles) -> 8 WMMAs per stage.
// Staging = pure b128 bf16 copies (no conversion in hot loop).
// Epilogue scatters into Q[B,H,T,D], K[B,H,T,D], Vt[B,H,D,T]  (bf16 bits).
// ---------------------------------------------------------------------------
__global__ __launch_bounds__(256) void qkv_gemm_kernel(
    const uint16_t* __restrict__ Xbf, const uint16_t* __restrict__ Wt,
    uint16_t* __restrict__ Qb, uint16_t* __restrict__ Kb, uint16_t* __restrict__ Vt)
{
  __shared__ __align__(16) uint16_t As[128 * 32];   // [m][k] row-major bf16
  __shared__ __align__(16) uint16_t Bs[128 * 32];   // [n][k] n-major bf16

  const int tid  = threadIdx.x;
  const int lane = tid & 31;
  const int wave = tid >> 5;
  const int wr = wave >> 2, wc = wave & 3;     // 2 x 4 wave grid
  const int m0 = blockIdx.y * 128;
  const int n0 = blockIdx.x * 128;
  const int l15 = lane & 15, lh = lane >> 4;

  FragF32 acc[4][2];
#pragma unroll
  for (int i = 0; i < 4; i++)
#pragma unroll
    for (int j = 0; j < 2; j++)
#pragma unroll
      for (int e = 0; e < 8; e++) acc[i][j].f[e] = 0.f;

  for (int k0 = 0; k0 < CC; k0 += 32) {
    // stage A & B: each 128x32 bf16 = 512 uint4, 2 per thread per side
#pragma unroll
    for (int i = 0; i < 2; i++) {
      int lin = tid + i * 256;
      int row = lin >> 2, c8 = (lin & 3) * 8;
      *(uint4*)&As[row * 32 + c8] =
          *(const uint4*)&Xbf[(size_t)(m0 + row) * CC + k0 + c8];
    }
#pragma unroll
    for (int i = 0; i < 2; i++) {
      int lin = tid + i * 256;
      int row = lin >> 2, c8 = (lin & 3) * 8;
      *(uint4*)&Bs[row * 32 + c8] =
          *(const uint4*)&Wt[(size_t)(n0 + row) * CC + k0 + c8];
    }
    // prefetch next K-slab into cache while this stage computes
    if (k0 + 32 < CC) {
      __builtin_prefetch(Xbf + (size_t)(m0 + (tid >> 1)) * CC + k0 + 32 + (tid & 1) * 16, 0, 0);
      __builtin_prefetch(Wt + (size_t)(n0 + (tid >> 1)) * CC + k0 + 32 + (tid & 1) * 16, 0, 0);
    }
    __syncthreads();

    FragB16 a[4], b[2];
#pragma unroll
    for (int t = 0; t < 4; t++) {
      int m = wr * 64 + t * 16 + l15;
#pragma unroll
      for (int j = 0; j < 8; j++) {
        int k = 2 * j + lh * 16;     // A-frag map
        a[t].u[j] = *(const uint32_t*)&As[m * 32 + k];
      }
    }
#pragma unroll
    for (int t = 0; t < 2; t++) {
      int n = wc * 32 + t * 16 + l15;
#pragma unroll
      for (int j = 0; j < 8; j++) {
        int k = 2 * j + lh * 16;                              // B-frag map
        b[t].u[j] = *(const uint32_t*)&Bs[n * 32 + k];
      }
    }
#pragma unroll
    for (int ti = 0; ti < 4; ti++)
#pragma unroll
      for (int tj = 0; tj < 2; tj++)
        acc[ti][tj].v = wmma_bf16(a[ti], b[tj], acc[ti][tj].v);
    __syncthreads();
  }

  // Epilogue: scatter to Q / K / Vt (bf16)
#pragma unroll
  for (int ti = 0; ti < 4; ti++) {
    int rb = m0 + wr * 64 + ti * 16 + lh * 8;   // row base for this lane-half
    int bidx = rb >> 11;                        // T = 2048, tiles never cross b
#pragma unroll
    for (int tj = 0; tj < 2; tj++) {
      int col  = n0 + wc * 32 + tj * 16 + l15;
      int part = col >> 10;          // 0=q 1=k 2=v (wave-uniform)
      int rem  = col & 1023;
      int head = rem >> 6;
      int d    = rem & 63;
#pragma unroll
      for (int j = 0; j < 8; j++) {
        int t = (rb + j) & (TT - 1);
        uint16_t v = f2bf(acc[ti][tj].f[j]);
        if (part == 0)      Qb[((size_t)(bidx * HH + head) * TT + t) * DD + d] = v;
        else if (part == 1) Kb[((size_t)(bidx * HH + head) * TT + t) * DD + d] = v;
        else                Vt[((size_t)(bidx * HH + head) * DD + d) * TT + t] = v;
      }
    }
  }
}

// ---------------------------------------------------------------------------
// Kernel 2: causal flash attention, one wave per 16-query tile.
// S = Q K^T (2 chained WMMA per 16-key tile), online softmax, O += P V.
// ---------------------------------------------------------------------------
__global__ __launch_bounds__(32) void attn_kernel(
    const uint16_t* __restrict__ Qb, const uint16_t* __restrict__ Kb,
    const uint16_t* __restrict__ Vt, uint16_t* __restrict__ Ob /* [B,T,C] bf16 */)
{
  __shared__ uint16_t Pbuf[16 * 32];   // P tile, row-major [query][key]

  const int lane = threadIdx.x;
  const int l15 = lane & 15, lh = lane >> 4;
  const int qt = blockIdx.x & 127;     // T/16 = 128
  const int bh = blockIdx.x >> 7;      // b*H + h
  const int b = bh >> 4, h = bh & 15;
  const int q0 = qt * 16;

  // Q A-fragments for both D-chunks (loaded once)
  FragB16 qf[2];
  {
    const uint16_t* qbase = Qb + ((size_t)bh * TT + (q0 + l15)) * DD;
#pragma unroll
    for (int c = 0; c < 2; c++)
#pragma unroll
      for (int j = 0; j < 8; j++) {
        int d = c * 32 + 2 * j + lh * 16;
        qf[c].u[j] = *(const uint32_t*)(qbase + d);
      }
  }

  float mrun[8], lrun[8];
  FragF32 Oacc[4];
#pragma unroll
  for (int j = 0; j < 8; j++) { mrun[j] = -3.0e38f; lrun[j] = 0.f; }
#pragma unroll
  for (int t = 0; t < 4; t++)
#pragma unroll
    for (int j = 0; j < 8; j++) Oacc[t].f[j] = 0.f;

  const float scale = 0.125f;                  // 1/sqrt(64)
  const int nkb = (q0 + 16 + 31) >> 5;         // # of 32-key blocks

  for (int kbi = 0; kbi < nkb; kbi++) {
    const int kb = kbi * 32;

    FragF32 S0, S1;
#pragma unroll
    for (int j = 0; j < 8; j++) { S0.f[j] = 0.f; S1.f[j] = 0.f; }

#pragma unroll
    for (int half = 0; half < 2; half++) {
      const uint16_t* kbase = Kb + ((size_t)bh * TT + (kb + half * 16 + l15)) * DD;
#pragma unroll
      for (int c = 0; c < 2; c++) {
        FragB16 kf;
#pragma unroll
        for (int j = 0; j < 8; j++) {
          int d = c * 32 + 2 * j + lh * 16;    // B-frag map over D
          kf.u[j] = *(const uint32_t*)(kbase + d);
        }
        if (half == 0) S0.v = wmma_bf16(qf[c], kf, S0.v);
        else           S1.v = wmma_bf16(qf[c], kf, S1.v);
      }
    }

    // online softmax over the 8 rows owned by this lane-half
#pragma unroll
    for (int j = 0; j < 8; j++) {
      int qrow = q0 + j + lh * 8;
      int key0 = kb + l15;
      float s0 = (key0 <= qrow)      ? S0.f[j] * scale : -3.0e38f;
      float s1 = (key0 + 16 <= qrow) ? S1.f[j] * scale : -3.0e38f;
      float mx = fmaxf(s0, s1);
      mx = fmaxf(mx, __shfl_xor(mx, 1, 32));
      mx = fmaxf(mx, __shfl_xor(mx, 2, 32));
      mx = fmaxf(mx, __shfl_xor(mx, 4, 32));
      mx = fmaxf(mx, __shfl_xor(mx, 8, 32));
      float newm  = fmaxf(mrun[j], mx);
      float alpha = __expf(mrun[j] - newm);
      float p0 = __expf(s0 - newm);
      float p1 = __expf(s1 - newm);
      float rs = p0 + p1;
      rs += __shfl_xor(rs, 1, 32);
      rs += __shfl_xor(rs, 2, 32);
      rs += __shfl_xor(rs, 4, 32);
      rs += __shfl_xor(rs, 8, 32);
      lrun[j] = lrun[j] * alpha + rs;
      mrun[j] = newm;
#pragma unroll
      for (int t = 0; t < 4; t++) Oacc[t].f[j] *= alpha;
      int prow = j + lh * 8;
      Pbuf[prow * 32 + l15]      = f2bf(p0);
      Pbuf[prow * 32 + 16 + l15] = f2bf(p1);
    }
    __syncthreads();   // D-layout -> A-layout via LDS

    FragB16 pf;
#pragma unroll
    for (int j = 0; j < 8; j++) {
      int k = 2 * j + lh * 16;   // A-frag map
      pf.u[j] = *(const uint32_t*)&Pbuf[l15 * 32 + k];
    }
#pragma unroll
    for (int t = 0; t < 4; t++) {
      const uint16_t* vbase = Vt + ((size_t)bh * DD + (t * 16 + l15)) * TT + kb;
      FragB16 vf;
#pragma unroll
      for (int j = 0; j < 8; j++) {
        int k = 2 * j + lh * 16;                          // B-frag map over keys
        vf.u[j] = *(const uint32_t*)(vbase + k);
      }
      Oacc[t].v = wmma_bf16(pf, vf, Oacc[t].v);
    }
    __syncthreads();
  }

  // epilogue: O /= l, write bf16 [B,T,C]
#pragma unroll
  for (int t = 0; t < 4; t++)
#pragma unroll
    for (int j = 0; j < 8; j++) {
      int r = j + lh * 8;
      int col = h * DD + t * 16 + l15;
      Ob[((size_t)b * TT + q0 + r) * CC + col] = f2bf(Oacc[t].f[j] / lrun[j]);
    }
}

// ---------------------------------------------------------------------------
// Kernel 3: output projection.  out[8192,1024] = Ab[8192,1024] @ Wpt[1024,1024]^T
// Same 128x128 / 8-wave tiling as kernel 1; all-bf16 staging.
// ---------------------------------------------------------------------------
__global__ __launch_bounds__(256) void proj_gemm_kernel(
    const uint16_t* __restrict__ Abf, const uint16_t* __restrict__ Wpt,
    float* __restrict__ Out)
{
  __shared__ __align__(16) uint16_t As[128 * 32];
  __shared__ __align__(16) uint16_t Bs[128 * 32];

  const int tid  = threadIdx.x;
  const int lane = tid & 31;
  const int wave = tid >> 5;
  const int wr = wave >> 2, wc = wave & 3;
  const int m0 = blockIdx.y * 128;
  const int n0 = blockIdx.x * 128;
  const int l15 = lane & 15, lh = lane >> 4;

  FragF32 acc[4][2];
#pragma unroll
  for (int i = 0; i < 4; i++)
#pragma unroll
    for (int j = 0; j < 2; j++)
#pragma unroll
      for (int e = 0; e < 8; e++) acc[i][j].f[e] = 0.f;

  for (int k0 = 0; k0 < CC; k0 += 32) {
#pragma unroll
    for (int i = 0; i < 2; i++) {
      int lin = tid + i * 256;
      int row = lin >> 2, c8 = (lin & 3) * 8;
      *(uint4*)&As[row * 32 + c8] =
          *(const uint4*)&Abf[(size_t)(m0 + row) * CC + k0 + c8];
    }
#pragma unroll
    for (int i = 0; i < 2; i++) {
      int lin = tid + i * 256;
      int row = lin >> 2, c8 = (lin & 3) * 8;
      *(uint4*)&Bs[row * 32 + c8] =
          *(const uint4*)&Wpt[(size_t)(n0 + row) * CC + k0 + c8];
    }
    if (k0 + 32 < CC) {
      __builtin_prefetch(Abf + (size_t)(m0 + (tid >> 1)) * CC + k0 + 32 + (tid & 1) * 16, 0, 0);
      __builtin_prefetch(Wpt + (size_t)(n0 + (tid >> 1)) * CC + k0 + 32 + (tid & 1) * 16, 0, 0);
    }
    __syncthreads();

    FragB16 a[4], b[2];
#pragma unroll
    for (int t = 0; t < 4; t++) {
      int m = wr * 64 + t * 16 + l15;
#pragma unroll
      for (int j = 0; j < 8; j++) {
        int k = 2 * j + lh * 16;
        a[t].u[j] = *(const uint32_t*)&As[m * 32 + k];
      }
    }
#pragma unroll
    for (int t = 0; t < 2; t++) {
      int n = wc * 32 + t * 16 + l15;
#pragma unroll
      for (int j = 0; j < 8; j++) {
        int k = 2 * j + lh * 16;
        b[t].u[j] = *(const uint32_t*)&Bs[n * 32 + k];
      }
    }
#pragma unroll
    for (int ti = 0; ti < 4; ti++)
#pragma unroll
      for (int tj = 0; tj < 2; tj++)
        acc[ti][tj].v = wmma_bf16(a[ti], b[tj], acc[ti][tj].v);
    __syncthreads();
  }

#pragma unroll
  for (int ti = 0; ti < 4; ti++) {
    int rb = m0 + wr * 64 + ti * 16 + lh * 8;
#pragma unroll
    for (int tj = 0; tj < 2; tj++) {
      int col = n0 + wc * 32 + tj * 16 + l15;
#pragma unroll
      for (int j = 0; j < 8; j++)
        Out[(size_t)(rb + j) * CC + col] = acc[ti][tj].f[j];
    }
  }
}

// ---------------------------------------------------------------------------
extern "C" void kernel_launch(void* const* d_in, const int* in_sizes, int n_in,
                              void* d_out, int out_size, void* d_ws, size_t ws_size,
                              hipStream_t stream) {
  const float* x     = (const float*)d_in[0];   // [B,T,C]
  const float* wqkv  = (const float*)d_in[1];   // [C,3C]
  const float* wproj = (const float*)d_in[2];   // [C,C]
  float* out = (float*)d_out;                   // [B,T,C]

  const size_t PER = (size_t)BB * HH * TT * DD; // 8,388,608 elements
  uint16_t* Qb    = (uint16_t*)d_ws;
  uint16_t* Kb    = Qb + PER;
  uint16_t* Vt    = Kb + PER;
  uint16_t* Ab    = Vt + PER;                   // attn out bf16 [B,T,C]
  uint16_t* Xb    = Ab + PER;                   // x bf16 [8192,1024]
  uint16_t* Wqt   = Xb + PER;                   // w_qkv^T bf16 [3072,1024]
  uint16_t* Wpt   = Wqt + (size_t)N_QKV * CC;   // w_proj^T bf16 [1024,1024]

  // One-time precision/layout prep
  convert_bf16_kernel<<<(MROWS * CC / 4 + 255) / 256, 256, 0, stream>>>(
      x, Xb, MROWS * CC / 4);
  dim3 gt1(N_QKV / 32, CC / 32);
  transpose_convert_kernel<<<gt1, 256, 0, stream>>>(wqkv, Wqt, CC, N_QKV);
  dim3 gt2(CC / 32, CC / 32);
  transpose_convert_kernel<<<gt2, 256, 0, stream>>>(wproj, Wpt, CC, CC);

  dim3 g1(N_QKV / 128, MROWS / 128);            // (24, 64)
  qkv_gemm_kernel<<<g1, 256, 0, stream>>>(Xb, Wqt, Qb, Kb, Vt);

  attn_kernel<<<BB * HH * (TT / 16), 32, 0, stream>>>(Qb, Kb, Vt, Ab);

  dim3 g2(CC / 128, MROWS / 128);               // (8, 64)
  proj_gemm_kernel<<<g2, 256, 0, stream>>>(Ab, Wpt, out);
}
